// AttentionModel_75247827026535
// MI455X (gfx1250) — matrix operand order repaired and checked by reading, not compiled, not run on hardware
//
#include <hip/hip_runtime.h>
#include <math.h>

// ---------------- types ----------------
typedef __attribute__((ext_vector_type(16))) __bf16 v16bf;
typedef __attribute__((ext_vector_type(8)))  __bf16 v8bf;
typedef __attribute__((ext_vector_type(4)))  __bf16 v4bf;
typedef __attribute__((ext_vector_type(8)))  float  v8f;
typedef __attribute__((ext_vector_type(4)))  float  v4f;

#define B_TOT   32
#define S_TOT   2048
#define D_TOT   1024
#define C_TOT   256
#define S_TILE  128
#define C_CHUNK 16
#define ROW     1032                 // padded LDS row length in bf16 elements (1024 + 8)
#define NEGV    1000000000000.0f

// round-to-nearest-even f32 -> bf16
static __device__ inline __bf16 f2bf(float f) {
    unsigned u = __builtin_bit_cast(unsigned, f);
    u += 0x7FFFu + ((u >> 16) & 1u);
    unsigned short h = (unsigned short)(u >> 16);
    return __builtin_bit_cast(__bf16, h);
}

// ---------------- k0: latA = (latent * att_diag) as bf16 ----------------
__global__ void k_latprep(const float* __restrict__ latent,
                          const float* __restrict__ att_diag,
                          unsigned short* __restrict__ latA_bits) {
    int e4  = blockIdx.x * blockDim.x + threadIdx.x;   // one float4 per thread
    int off = e4 * 4;                                  // covers C_TOT*D_TOT
    int d   = off & (D_TOT - 1);
    v4f l = *(const v4f*)(latent + off);
    v4f a = *(const v4f*)(att_diag + d);
    v4bf o;
#pragma unroll
    for (int j = 0; j < 4; ++j) o[j] = f2bf(l[j] * a[j]);
    *(v4bf*)((__bf16*)latA_bits + off) = o;
}

// ---------------- k1: scores GEMM (bf16 WMMA) + masked max over C ----------------
__global__ void __launch_bounds__(256, 1)
k_scores_max(const float* __restrict__ embeds,
             const float* __restrict__ embeds_mask,
             const unsigned short* __restrict__ latA_bits,
             const float* __restrict__ pos_table,
             const float* __restrict__ tok_mult,
             const int* __restrict__ rel_pos_ids,
             float* __restrict__ top) {
    extern __shared__ char smem[];
    __bf16* sM   = (__bf16*)smem;                                  // [S_TILE][ROW]
    __bf16* sA   = (__bf16*)(smem + S_TILE * ROW * 2);             // [C_CHUNK][ROW]
    int*    sPos = (int*)   (smem + (S_TILE + C_CHUNK) * ROW * 2); // [S_TILE]
    const __bf16* latA = (const __bf16*)latA_bits;

    const int tid = threadIdx.x;
    const int b   = blockIdx.x >> 4;            // 16 s-tiles per batch
    const int st  = (blockIdx.x & 15) * S_TILE;
    const float tm = tok_mult[0];

    if (tid < S_TILE) sPos[tid] = rel_pos_ids[b * S_TOT + st + tid] + 64;
    __syncthreads();

    // stage merged tile (tok_mult*embeds + pos_embed) as bf16, row-major [s][d]
    for (int idx = tid; idx < S_TILE * (D_TOT / 4); idx += 256) {
        int s  = idx >> 8;                      // D_TOT/4 = 256 float4 per row
        int d4 = (idx & 255) << 2;
        v4f e = *(const v4f*)(embeds + (size_t)(b * S_TOT + st + s) * D_TOT + d4);
        v4f p = *(const v4f*)(pos_table + (size_t)sPos[s] * D_TOT + d4);
        v4bf o;
#pragma unroll
        for (int j = 0; j < 4; ++j) o[j] = f2bf(tm * e[j] + p[j]);
        *(v4bf*)(sM + s * ROW + d4) = o;
    }

    const int wave = tid >> 5;
    const int lane = tid & 31;
    const int l    = lane & 15;
    const int hi   = lane >> 4;                 // 0: K {0..7,16..23}, 1: K {8..15,24..31} of each 32-step
    const int srow = wave * 16 + l;             // B-fragment row in sM

    // LDS byte offset of sA (dynamic LDS region starts at workgroup offset 0;
    // this kernel declares no static __shared__)
    const unsigned sA_lds_base = (unsigned)(S_TILE * ROW * 2);

    float tmax = -INFINITY;

    for (int c0 = 0; c0 < C_TOT; c0 += C_CHUNK) {
        __syncthreads();
        // stage 16 rows of latA straight from L2 into LDS with the CDNA5
        // async-to-LDS path (16 B per lane, tracked by ASYNCcnt)
        for (int g = tid; g < C_CHUNK * (D_TOT / 8); g += 256) {
            int c  = g >> 7;                    // D_TOT/8 = 128 16B-groups per row
            int dd = (g & 127) << 3;
            unsigned lds = sA_lds_base + (unsigned)(c * ROW + dd) * 2u;
            const __bf16* src = latA + (size_t)(c0 + c) * D_TOT + dd;
            asm volatile("global_load_async_to_lds_b128 %0, %1, off"
                         :: "v"(lds), "v"(src)
                         : "memory");
        }
        asm volatile("s_wait_asynccnt 0x0" ::: "memory");
        __syncthreads();

        v8f acc;
#pragma unroll
        for (int j = 0; j < 8; ++j) acc[j] = 0.0f;

#pragma unroll 4
        for (int k = 0; k < D_TOT / 32; ++k) {
            const int d0 = k * 32;
            // A fragment (16x32 bf16): lane l holds row c0+l,
            //   elems 0..7  -> K = d0 + hi*8 + {0..7}
            //   elems 8..15 -> K = d0 + 16 + hi*8 + {0..7}
            const __bf16* pa = sA + l * ROW + d0 + hi * 8;
            v8bf a0 = *(const v8bf*)(pa);
            v8bf a1 = *(const v8bf*)(pa + 16);
            v16bf afrag = __builtin_shufflevector(a0, a1,
                0,1,2,3,4,5,6,7,8,9,10,11,12,13,14,15);
            // B fragment (32x16 bf16): lane l holds col s, K = d0 + hi*8 + {0..7}, d0 + 16 + hi*8 + {0..7}
            const __bf16* pb = sM + srow * ROW + d0 + hi * 8;
            v8bf b0 = *(const v8bf*)(pb);
            v8bf b1 = *(const v8bf*)(pb + 16);
            v16bf bfrag = __builtin_shufflevector(b0, b1,
                0,1,2,3,4,5,6,7,8,9,10,11,12,13,14,15);

            acc = __builtin_amdgcn_wmma_f32_16x16x32_bf16(
                false, afrag, false, bfrag, (short)0, acc, false, false);
        }

        float mx = acc[0];
#pragma unroll
        for (int j = 1; j < 8; ++j) mx = fmaxf(mx, acc[j]);
        tmax = fmaxf(tmax, mx);
    }

    // lanes l and l+16 hold C rows {..+0..7} and {..+8..15} of the same column s
    float other = __shfl_xor(tmax, 16);
    tmax = fmaxf(tmax, other);
    if (hi == 0) {
        int s = st + wave * 16 + l;
        float m = embeds_mask[b * S_TOT + s];
        top[b * S_TOT + s] = tmax * m + (m - 1.0f) * NEGV;
    }
}

// ---------------- k2: per-batch softmax over S ----------------
__global__ void k_softmax(const float* __restrict__ top,
                          float* __restrict__ probs) {
    __shared__ float red[256];
    const int b = blockIdx.x, tid = threadIdx.x;
    const float* t = top + b * S_TOT;
    float* p = probs + b * S_TOT;

    float lmax = -INFINITY;
    for (int s = tid; s < S_TOT; s += 256) lmax = fmaxf(lmax, t[s]);
    red[tid] = lmax; __syncthreads();
    for (int off = 128; off > 0; off >>= 1) {
        if (tid < off) red[tid] = fmaxf(red[tid], red[tid + off]);
        __syncthreads();
    }
    const float gmax = red[0];
    __syncthreads();

    float lsum = 0.0f;
    for (int s = tid; s < S_TOT; s += 256) {
        float e = __expf(t[s] - gmax);
        p[s] = e;
        lsum += e;
    }
    red[tid] = lsum; __syncthreads();
    for (int off = 128; off > 0; off >>= 1) {
        if (tid < off) red[tid] += red[tid + off];
        __syncthreads();
    }
    const float inv = 1.0f / red[0];
    for (int s = tid; s < S_TOT; s += 256) p[s] *= inv;
}

// ---------------- k3: ctx[b,d] = tok_diag[d] * sum_s probs[b,s]*embeds[b,s,d] ----------------
__global__ void k_ctx(const float* __restrict__ embeds,
                      const float* __restrict__ tok_diag,
                      const float* __restrict__ probs,
                      float* __restrict__ out) {
    __shared__ float red[256];
    const int b  = blockIdx.x >> 3;
    const int d0 = (blockIdx.x & 7) * 128;
    const int tid = threadIdx.x;
    const int d  = d0 + (tid & 127);
    const int sp = tid >> 7;                    // s-parity split
    const float* pb = probs + b * S_TOT;

    float acc = 0.0f;
    for (int s = sp; s < S_TOT; s += 2)
        acc += pb[s] * embeds[(size_t)(b * S_TOT + s) * D_TOT + d];
    red[tid] = acc; __syncthreads();
    if (sp == 0)
        out[b * D_TOT + d] = (red[tid] + red[tid + 128]) * tok_diag[d];
}

// ---------------- host launcher ----------------
extern "C" void kernel_launch(void* const* d_in, const int* in_sizes, int n_in,
                              void* d_out, int out_size, void* d_ws, size_t ws_size,
                              hipStream_t stream) {
    const float* embeds   = (const float*)d_in[0];
    const float* mask     = (const float*)d_in[1];
    const float* latent   = (const float*)d_in[2];
    const float* att_diag = (const float*)d_in[3];
    const float* tok_diag = (const float*)d_in[4];
    const float* pos_tab  = (const float*)d_in[5];
    const float* tok_mult = (const float*)d_in[6];
    const int*   rel_ids  = (const int*)d_in[7];

    char* ws = (char*)d_ws;
    unsigned short* latA = (unsigned short*)ws;                 // 256*1024 bf16 = 512 KB
    float* top   = (float*)(ws + 524288);                       // 32*2048 f32  = 256 KB
    float* probs = (float*)(ws + 786432);                       // 32*2048 f32  = 256 KB

    // k0: 256*1024 elems / (256 thr * 4 per thr) = 256 blocks
    k_latprep<<<256, 256, 0, stream>>>(latent, att_diag, latA);

    // k1: B * (S/S_TILE) = 32*16 = 512 blocks; dynamic LDS:
    //   merged 128*1032*2 + latA chunk 16*1032*2 + 128 ints = 297728 B
    const size_t smem = (size_t)(S_TILE + C_CHUNK) * ROW * 2 + S_TILE * 4;
    k_scores_max<<<512, 256, smem, stream>>>(embeds, mask, latA, pos_tab,
                                             tok_mult, rel_ids, top);

    // k2: one block per batch
    k_softmax<<<B_TOT, 256, 0, stream>>>(top, probs);

    // k3: B * (D/128) = 256 blocks
    k_ctx<<<256, 256, 0, stream>>>(embeds, tok_diag, probs, (float*)d_out);
}
